// MultiQueryAttention_73280732005042
// MI455X (gfx1250) — matrix operand order repaired and checked by reading, not compiled, not run on hardware
//
#include <hip/hip_runtime.h>

// ---------------------------------------------------------------------------
// MQA forward for gfx1250 (CDNA5 / MI455X), wave32 + WMMA.
//   out    = (softmax(mask(QK^T*s)) V) @ Wo + bo    [1024 x 2048] fp32
//   scores = softmax probabilities                   [256 x 1024 x 1024] fp32
// d_out = [out | scores].  Memory-bound on the 1 GiB scores write: scores are
// produced exactly once with non-temporal stores; the two-pass online softmax
// recomputes QK^T on the matrix units instead of re-reading scores.
// P@V also runs on WMMA via a per-wave LDS transpose of the P tile.
// GEMMs: both A and B tiles stream via GLOBAL_LOAD_ASYNC_TO_LDS_B128
// (ASYNCcnt), double-buffered; inner loop: A via ds_load_b128, B via strided 16-bit LDS reads -> v_wmma.
// ---------------------------------------------------------------------------

typedef _Float16 v16h __attribute__((ext_vector_type(16)));
typedef _Float16 v8h  __attribute__((ext_vector_type(8)));
typedef _Float16 v4h  __attribute__((ext_vector_type(4)));
typedef float    v8f  __attribute__((ext_vector_type(8)));
typedef float    v2f  __attribute__((ext_vector_type(2)));

// async-copy builtin operand types: V4i*1 (global), V4i*3 (LDS)
typedef int b128v __attribute__((__vector_size__(16)));
typedef b128v __attribute__((address_space(1)))* b128_gp;
typedef b128v __attribute__((address_space(3)))* b128_lp;

constexpr int   S_DIM    = 1024;
constexpr int   D_MODEL  = 2048;
constexpr int   N_HEADS  = 256;
constexpr int   HEAD_DIM = 8;
constexpr float ATT_SCALE = 0.35355339059327379f;   // 1/sqrt(8)

#if defined(__AMDGCN__) && __has_builtin(__builtin_amdgcn_global_load_async_to_lds_b128)
#define USE_ASYNC_COPY 1
#else
#define USE_ASYNC_COPY 0
#endif

__device__ __forceinline__ void async_wait() {
#if USE_ASYNC_COPY
#if __has_builtin(__builtin_amdgcn_s_wait_asynccnt)
  __builtin_amdgcn_s_wait_asynccnt(0);
#else
  asm volatile("s_wait_asynccnt 0" ::: "memory");
#endif
#endif
}

__device__ __forceinline__ void async_copy16(const _Float16* g, _Float16* l) {
#if USE_ASYNC_COPY
  __builtin_amdgcn_global_load_async_to_lds_b128(
      (b128_gp)(__UINTPTR_TYPE__)g, (b128_lp)(unsigned)(__UINTPTR_TYPE__)l, 0, 0);
#else
  *(v8h*)l = *(const v8h*)g;
#endif
}

// ---------------------------------------------------------------------------
// fp32 -> f16 conversion (x and weights), 4-wide
// ---------------------------------------------------------------------------
__global__ void cvt_f32_to_f16(const float* __restrict__ src,
                               _Float16* __restrict__ dst, int n) {
  int i      = (blockIdx.x * blockDim.x + threadIdx.x) * 4;
  int stride = gridDim.x * blockDim.x * 4;
  for (; i < n; i += stride) {
    float4 v = *(const float4*)(src + i);
    v4h o = { (_Float16)v.x, (_Float16)v.y, (_Float16)v.z, (_Float16)v.w };
    *(v4h*)(dst + i) = o;
  }
}

// ---------------------------------------------------------------------------
// KV projection: KV[m][0..7]=K, KV[m][8..15]=V.  M=1024, N=16, K=2048.
// One wave per 16-row tile; f16 WMMA, A from pre-converted xh (b128 loads).
// ---------------------------------------------------------------------------
__global__ void kv_proj_kernel(const _Float16* __restrict__ xh,
                               const float* __restrict__ Wk,
                               const float* __restrict__ bk,
                               const float* __restrict__ Wv,
                               const float* __restrict__ bv,
                               float* __restrict__ KV) {
  const int wave = threadIdx.x >> 5;
  const int lane = threadIdx.x & 31;
  const int hi   = lane >> 4;
  const int ln   = lane & 15;
  const int m0   = (blockIdx.x * 8 + wave) * 16;
  const int mrow = m0 + ln;
  const int kbA  = hi * 8;           // A frag: K = kbA+{0..7}, kbA+16+{0..7}
  const int kbB  = hi * 8;           // B frag: K = kbB+{0..7}, kbB+16+{0..7}

  const float* Wcol = (ln < 8) ? (Wk + ln) : (Wv + (ln - 8));
  const float  bb   = (ln < 8) ? bk[ln] : bv[ln - 8];

  v8f c = {};
  for (int k0 = 0; k0 < D_MODEL; k0 += 32) {
    v16h a, b;
    const _Float16* xr = xh + (size_t)mrow * D_MODEL + k0 + kbA;
    v8h lo = *(const v8h*)xr;
    v8h hi8 = *(const v8h*)(xr + 16);
#pragma unroll
    for (int i = 0; i < 8; ++i) { a[i] = lo[i]; a[8 + i] = hi8[i]; }
#pragma unroll
    for (int i = 0; i < 16; ++i) b[i] = (_Float16)Wcol[(size_t)(k0 + kbB + (i & 7) + (i >> 3) * 16) * HEAD_DIM];
    c = __builtin_amdgcn_wmma_f32_16x16x32_f16(false, a, false, b, (short)0, c,
                                               false, false);
  }
#pragma unroll
  for (int r = 0; r < 8; ++r) {
    int m = m0 + r + hi * 8;
    KV[m * 16 + ln] = c[r] + bb;
  }
}

// ---------------------------------------------------------------------------
// Tiled GEMM: C[1024x2048](f32) = Ah(f16)[1024x2048] @ Bh(f16)[2048x2048]+bias
// Block 128x128, BK=32, 8 waves (2x4) of 64x32 wave tiles.  Both operand
// tiles stream via async global->LDS b128, double-buffered.
// ---------------------------------------------------------------------------
__global__ void gemm_f16_128(const _Float16* __restrict__ Ah,
                             const _Float16* __restrict__ Bh,
                             const float* __restrict__ bias,
                             float* __restrict__ Cout) {
  constexpr int LDA = 40;            // A row stride (halves): 80 B = 5*16
  constexpr int LDB = 136;           // B row stride (halves): 272 B = 17*16
  __shared__ __align__(16) _Float16 As[2][128 * LDA];  // [m][k]
  __shared__ __align__(16) _Float16 Bs[2][32 * LDB];   // [k][n]

  const int tid  = threadIdx.x;
  const int wave = tid >> 5;
  const int lane = tid & 31;
  const int hi   = lane >> 4;
  const int ln   = lane & 15;
  const int wm   = wave >> 2;
  const int wn   = wave & 3;
  const int mBase = blockIdx.y * 128;
  const int nBase = blockIdx.x * 128;
  const int row0  = wm * 64;
  const int col0  = wn * 32;

  v8f c[4][2];
#pragma unroll
  for (int mi = 0; mi < 4; ++mi)
#pragma unroll
    for (int ni = 0; ni < 2; ++ni) c[mi][ni] = v8f{};

  // 512 16B chunks per tile; thread t moves chunks t and t+256 of each tile
  auto stage = [&](int k0, int buf) {
#pragma unroll
    for (int ch = 0; ch < 2; ++ch) {
      const int chunk = tid + ch * 256;
      // A tile 128x32 halves: 4 chunks per row
      {
        const int r  = chunk >> 2;
        const int kb = (chunk & 3) * 8;
        async_copy16(Ah + (size_t)(mBase + r) * D_MODEL + k0 + kb,
                     &As[buf][r * LDA + kb]);
      }
      // B tile 32x128 halves: 16 chunks per row
      {
        const int kk = chunk >> 4;
        const int nb = (chunk & 15) * 8;
        async_copy16(Bh + (size_t)(k0 + kk) * D_MODEL + nBase + nb,
                     &Bs[buf][kk * LDB + nb]);
      }
    }
  };

  stage(0, 0);
  async_wait();
  __syncthreads();

  constexpr int NT = D_MODEL / 32;
  for (int kt = 0; kt < NT; ++kt) {
    const int buf = kt & 1;
    if (kt + 1 < NT) stage((kt + 1) * 32, buf ^ 1);

    v16h af[4], bf[2];
#pragma unroll
    for (int mi = 0; mi < 4; ++mi) {
      const _Float16* rp = &As[buf][(row0 + mi * 16 + ln) * LDA + hi * 8];
      v8h lo = *(const v8h*)rp;
      v8h hi8 = *(const v8h*)(rp + 16);
#pragma unroll
      for (int i = 0; i < 8; ++i) { af[mi][i] = lo[i]; af[mi][8 + i] = hi8[i]; }
    }
#pragma unroll
    for (int ni = 0; ni < 2; ++ni) {
      const _Float16* bp = &Bs[buf][(hi * 8) * LDB + col0 + ni * 16 + ln];
#pragma unroll
      for (int i = 0; i < 16; ++i) bf[ni][i] = bp[((i & 7) + (i >> 3) * 16) * LDB];
    }
#pragma unroll
    for (int mi = 0; mi < 4; ++mi)
#pragma unroll
      for (int ni = 0; ni < 2; ++ni)
        c[mi][ni] = __builtin_amdgcn_wmma_f32_16x16x32_f16(
            false, af[mi], false, bf[ni], (short)0, c[mi][ni], false, false);

    async_wait();
    __syncthreads();
  }

#pragma unroll
  for (int ni = 0; ni < 2; ++ni) {
    const int nn = nBase + col0 + ni * 16 + ln;
    const float bb = bias[nn];
#pragma unroll
    for (int mi = 0; mi < 4; ++mi)
#pragma unroll
      for (int r = 0; r < 8; ++r) {
        int m = mBase + row0 + mi * 16 + r + hi * 8;
        Cout[(size_t)m * D_MODEL + nn] = c[mi][ni][r] + bb;
      }
  }
}

// ---------------------------------------------------------------------------
// QK^T 16x16 tile via fp32 WMMA (head_dim 8 = two K=4 steps).
// KV row layout: [j][0..7]=K, [j][8..15]=V.
// ---------------------------------------------------------------------------
__device__ __forceinline__ v8f qk_tile(v2f a0, v2f a1,
                                       const float* __restrict__ KV,
                                       int j0, int ln, int dA) {
  const float* kp = KV + (j0 + ln) * 16 + dA;
  v2f b0 = { kp[0], kp[1] };
  v2f b1 = { kp[4], kp[5] };
  v8f c = {};
  c = __builtin_amdgcn_wmma_f32_16x16x4_f32(false, a1, false, b1, (short)0, c,
                                            false, false);
  c = __builtin_amdgcn_wmma_f32_16x16x4_f32(false, a0, false, b0, (short)0, c,
                                            false, false);
  return c;
}

// ---------------------------------------------------------------------------
// Attention: per (head, 128-q block); each wave owns 16 q rows.
// Pass 1: online (max, sumexp) per row via WMMA-recomputed QK^T.
// Pass 2: recompute, non-temporal scores store (zeros in masked triangle),
//         P@V on WMMA via per-wave LDS transpose of the P tile.
// Output o_heads written as f16 (feeds the f16 output GEMM directly).
// ---------------------------------------------------------------------------
__global__ void attn_kernel(const float* __restrict__ Qbuf,
                            const float* __restrict__ KV,
                            float* __restrict__ scores,
                            _Float16* __restrict__ o_heads) {
  constexpr int PLD = 18;            // P scratch row stride (72 B: 8B-aligned)
  __shared__ __align__(16) float Pl[8][16][PLD];

  const int h    = blockIdx.y;
  const int wave = threadIdx.x >> 5;
  const int lane = threadIdx.x & 31;
  const int hi   = lane >> 4;
  const int ln   = lane & 15;
  const int q0   = blockIdx.x * 128 + wave * 16;
  const int dA   = hi * 2;           // fp32 WMMA frag K-pair base
  float (*pw)[PLD] = Pl[wave];

  const float* qp = Qbuf + (size_t)(q0 + ln) * D_MODEL + h * HEAD_DIM;
  v2f a0 = { qp[dA],     qp[dA + 1] };
  v2f a1 = { qp[4 + dA], qp[5 + dA] };

  const int jtMax = (q0 >> 4) + 1;   // causal: tiles with any j <= q0+15

  // ---- pass 1 ----
  float mrow[8], srow[8];
#pragma unroll
  for (int r = 0; r < 8; ++r) { mrow[r] = -1e30f; srow[r] = 0.f; }

  for (int jt = 0; jt < jtMax; ++jt) {
    const int j0 = jt * 16;
    v8f c = qk_tile(a0, a1, KV, j0, ln, dA);
    const int j = j0 + ln;
#pragma unroll
    for (int r = 0; r < 8; ++r) {
      int q = q0 + r + hi * 8;
      if (j <= q) {
        float s  = c[r] * ATT_SCALE;
        float nm = fmaxf(mrow[r], s);
        srow[r]  = srow[r] * __expf(mrow[r] - nm) + __expf(s - nm);
        mrow[r]  = nm;
      }
    }
  }
#pragma unroll
  for (int r = 0; r < 8; ++r) {
    float m = mrow[r], s = srow[r];
#pragma unroll
    for (int off = 1; off < 16; off <<= 1) {
      float om = __shfl_xor(m, off, 32);
      float os = __shfl_xor(s, off, 32);
      float M  = fmaxf(m, om);
      s = s * __expf(m - M) + os * __expf(om - M);
      m = M;
    }
    mrow[r] = m;
    srow[r] = 1.0f / s;
  }

  // ---- pass 2 ----
  v8f oc = {};                        // P@V accumulator: rows q, cols d (0..7)
  for (int jt = 0; jt < S_DIM / 16; ++jt) {
    const int j0 = jt * 16;
    const int j  = j0 + ln;
    const bool active = (j0 < q0 + 16);   // wave-uniform
    float p[8];
    if (active) {
      v8f c = qk_tile(a0, a1, KV, j0, ln, dA);
#pragma unroll
      for (int r = 0; r < 8; ++r) {
        int q = q0 + r + hi * 8;
        p[r] = (j <= q) ? __expf(c[r] * ATT_SCALE - mrow[r]) * srow[r] : 0.f;
      }
    } else {
#pragma unroll
      for (int r = 0; r < 8; ++r) p[r] = 0.f;
    }
    // single NT write of the 1 GiB scores tensor (64 B/row coalesced)
#pragma unroll
    for (int r = 0; r < 8; ++r) {
      int q = q0 + r + hi * 8;
      __builtin_nontemporal_store(p[r], &scores[((size_t)h * S_DIM + q) * S_DIM + j]);
    }
    if (active) {
      // C-layout -> A-layout via per-wave LDS scratch, then 4x f32 WMMA
#pragma unroll
      for (int r = 0; r < 8; ++r) pw[r + hi * 8][ln] = p[r];
#pragma unroll
      for (int t = 0; t < 4; ++t) {
        v2f aP = { pw[ln][4 * t + dA], pw[ln][4 * t + dA + 1] };
        v2f bV;
        if (ln < 8) {
          bV[0] = KV[(j0 + 4 * t + dA)     * 16 + 8 + ln];
          bV[1] = KV[(j0 + 4 * t + dA + 1) * 16 + 8 + ln];
        } else { bV[0] = 0.f; bV[1] = 0.f; }
        oc = __builtin_amdgcn_wmma_f32_16x16x4_f32(false, aP, false, bV,
                                                   (short)0, oc, false, false);
      }
    }
  }

  if (ln < 8) {
#pragma unroll
    for (int r = 0; r < 8; ++r) {
      int q = q0 + r + hi * 8;
      o_heads[(size_t)q * D_MODEL + h * HEAD_DIM + ln] = (_Float16)oc[r];
    }
  }
}

// ---------------------------------------------------------------------------
extern "C" void kernel_launch(void* const* d_in, const int* in_sizes, int n_in,
                              void* d_out, int out_size, void* d_ws, size_t ws_size,
                              hipStream_t stream) {
  (void)in_sizes; (void)n_in; (void)out_size; (void)ws_size;
  const float* x  = (const float*)d_in[0];
  // d_in[1] = causal mask (tril) — implicit in kernel
  const float* Wq = (const float*)d_in[2];
  const float* bq = (const float*)d_in[3];
  const float* Wk = (const float*)d_in[4];
  const float* bk = (const float*)d_in[5];
  const float* Wv = (const float*)d_in[6];
  const float* bv = (const float*)d_in[7];
  const float* Wo = (const float*)d_in[8];
  const float* bo = (const float*)d_in[9];

  float* out    = (float*)d_out;
  float* scores = out + (size_t)S_DIM * D_MODEL;   // tuple: (out, scores)

  char* ws = (char*)d_ws;
  _Float16* Wqh    = (_Float16*)(ws);                        // 8 MB
  _Float16* Woh    = (_Float16*)(ws + (size_t)8  * 1048576); // 8 MB
  _Float16* xh     = (_Float16*)(ws + (size_t)16 * 1048576); // 4 MB
  float*    Qbuf   = (float*)   (ws + (size_t)20 * 1048576); // 8 MB
  _Float16* o_head = (_Float16*)(ws + (size_t)28 * 1048576); // 4 MB
  float*    KVbuf  = (float*)   (ws + (size_t)32 * 1048576); // 64 KB

  const int nW = D_MODEL * D_MODEL;
  cvt_f32_to_f16<<<1024, 256, 0, stream>>>(Wq, Wqh, nW);
  cvt_f32_to_f16<<<1024, 256, 0, stream>>>(Wo, Woh, nW);
  cvt_f32_to_f16<<<1024, 256, 0, stream>>>(x, xh, S_DIM * D_MODEL);

  kv_proj_kernel<<<8, 256, 0, stream>>>(xh, Wk, bk, Wv, bv, KVbuf);

  gemm_f16_128<<<dim3(D_MODEL / 128, S_DIM / 128), 256, 0, stream>>>(xh, Wqh, bq,
                                                                     Qbuf);

  attn_kernel<<<dim3(S_DIM / 128, N_HEADS), 256, 0, stream>>>(Qbuf, KVbuf,
                                                              scores, o_head);

  gemm_f16_128<<<dim3(D_MODEL / 128, S_DIM / 128), 256, 0, stream>>>(o_head, Woh,
                                                                     bo, out);
}
